// SubexpertIntegration_18124761989360
// MI455X (gfx1250) — compile-verified, matrix operand order checked by reading, not run on hardware
//
#include <hip/hip_runtime.h>
#include <hip/hip_bf16.h>

// ---------------------------------------------------------------------------
// MoE subexpert integration for gfx1250 (MI455X), bf16 WMMA path.
//   out[b,n] = sum_e gate[b,e] * relu(bn2(relu(bn1(X@W1[e]+b1))@W2[e]+b2))
//
// Fully fused per 32-row block: expert loop inside the block, h staged in LDS,
// output written once (no atomics).  All matmuls use v_wmma_f32_16x16x32_bf16.
//
// All WMMA operands are pre-packed into fragment order ("swizzled"):
//   per (16-wide tile, 32-deep K step) one contiguous 1KB block =
//   plane0[lane 0..31][8 halves] ++ plane1[lane 0..31][8 halves]
// so every operand fetch is a dense, wave-contiguous global_load_b128 /
// ds_load_b128 (lane-linear, bank-conflict-free).
// ---------------------------------------------------------------------------

typedef __attribute__((ext_vector_type(16))) __bf16       v16bf;
typedef __attribute__((ext_vector_type(8)))  __bf16       v8bf;
typedef __attribute__((ext_vector_type(8)))  float        v8f;
typedef __attribute__((ext_vector_type(4)))  unsigned int v4u;

#define B_   16384
#define D_   2048
#define S_   512
#define E_   8
#define H1_  1024
#define H2_  512
#define EPS_ 1e-5f

#define BT 32          // batch rows per block
#define KS1 (D_ / 32)  // 64 K-steps in GEMM1
#define KS2 (H1_ / 32) // 32 K-steps in GEMM2

// Fragment block: 512 halves.  Lane's 16 elements = 8 at (base + lane*8),
// 8 at (base + 256 + lane*8).  Maps to the CDNA5 16-bit A/B VGPR layout.
__device__ __forceinline__ v16bf load_frag_g(const __bf16* base, int lane) {
    union { v4u q[2]; v16bf v; } u;
    u.q[0] = *(const v4u*)(base + lane * 8);
    u.q[1] = *(const v4u*)(base + lane * 8 + 256);
    return u.v;
}
__device__ __forceinline__ v16bf load_frag_l(const __bf16* base, int lane) {
    union { v4u q[2]; v16bf v; } u;
    u.q[0] = *(const v4u*)(base + lane * 8);
    u.q[1] = *(const v4u*)(base + lane * 8 + 256);
    return u.v;
}

// ---------------------------------------------------------------------------
// Packing kernels: f32 sources -> bf16 fragment-order layouts.
// One thread produces one 16B chunk (8 halves along K); writes are dense.
// ---------------------------------------------------------------------------

// X[B,D] f32 -> Xs[rb = b/16][ks = k/32][plane][lane][8] bf16
__global__ void k_pack_X(const float* __restrict__ X, __bf16* __restrict__ Xs) {
    int tid = blockIdx.x * 256 + threadIdx.x;
    int l  = tid & 31;
    int c  = (tid >> 5) & 1;
    int ks = (tid >> 6) & (KS1 - 1);
    int rb = tid >> 12;
    int m  = rb * 16 + (l & 15);
    int kb = ks * 32 + c * 16 + (l >> 4) * 8;
    const float* src = X + (long)m * D_ + kb;
    v8bf o;
#pragma unroll
    for (int j = 0; j < 8; ++j) o[j] = (__bf16)src[j];
    *(v8bf*)(Xs + ((long)(rb * KS1 + ks) * 512 + c * 256 + l * 8)) = o;
}

// W1[E,D,H1] f32 -> W1s[e][ntl = n/16][ks = k/32][plane][lane][8] bf16
__global__ void k_pack_W1(const float* __restrict__ W, __bf16* __restrict__ Ws) {
    int tid = blockIdx.x * 256 + threadIdx.x;
    int l   = tid & 31;
    int c   = (tid >> 5) & 1;
    int ks  = (tid >> 6) & (KS1 - 1);
    int ntl = (tid >> 12) & (H1_ / 16 - 1);
    int e   = tid >> 18;
    int n   = ntl * 16 + (l & 15);
    int kb  = ks * 32 + c * 16 + (l >> 4) * 8;
    const float* src = W + (long)e * D_ * H1_ + (long)kb * H1_ + n;
    v8bf o;
#pragma unroll
    for (int j = 0; j < 8; ++j) o[j] = (__bf16)src[(long)j * H1_];
    *(v8bf*)(Ws + ((long)((e * (H1_ / 16) + ntl) * KS1 + ks) * 512 + c * 256 + l * 8)) = o;
}

// W2[E,H1,H2] f32 -> W2s[e][ntl = n/16][ks = k/32][plane][lane][8] bf16
__global__ void k_pack_W2(const float* __restrict__ W, __bf16* __restrict__ Ws) {
    int tid = blockIdx.x * 256 + threadIdx.x;
    int l   = tid & 31;
    int c   = (tid >> 5) & 1;
    int ks  = (tid >> 6) & (KS2 - 1);
    int ntl = (tid >> 11) & (H2_ / 16 - 1);
    int e   = tid >> 16;
    int n   = ntl * 16 + (l & 15);
    int kb  = ks * 32 + c * 16 + (l >> 4) * 8;
    const float* src = W + (long)e * H1_ * H2_ + (long)kb * H2_ + n;
    v8bf o;
#pragma unroll
    for (int j = 0; j < 8; ++j) o[j] = (__bf16)src[(long)j * H2_];
    *(v8bf*)(Ws + ((long)((e * (H2_ / 16) + ntl) * KS2 + ks) * 512 + c * 256 + l * 8)) = o;
}

// gate = softmax(sideinfo @ Wg + bg), one row per thread, Wg staged in LDS
__global__ void k_gate(const float* __restrict__ side,
                       const float* __restrict__ Wg,
                       const float* __restrict__ bg,
                       float* __restrict__ gate) {
    __shared__ float wg[S_ * E_];           // 16 KB
    for (int i = threadIdx.x; i < S_ * E_; i += 256) wg[i] = Wg[i];
    __syncthreads();
    int b = blockIdx.x * 256 + threadIdx.x;
    float acc[E_];
#pragma unroll
    for (int e = 0; e < E_; ++e) acc[e] = bg[e];
    const float* sr = side + (long)b * S_;
    for (int s = 0; s < S_; ++s) {
        float x = sr[s];
#pragma unroll
        for (int e = 0; e < E_; ++e) acc[e] = fmaf(x, wg[s * E_ + e], acc[e]);
    }
    float mx = acc[0];
#pragma unroll
    for (int e = 1; e < E_; ++e) mx = fmaxf(mx, acc[e]);
    float sum = 0.f;
#pragma unroll
    for (int e = 0; e < E_; ++e) { acc[e] = __expf(acc[e] - mx); sum += acc[e]; }
    float inv = 1.f / sum;
#pragma unroll
    for (int e = 0; e < E_; ++e) gate[(long)b * E_ + e] = acc[e] * inv;
}

// ---------------------------------------------------------------------------
// Fused expert kernel.  Block = 256 threads (8 waves), owns 32 batch rows.
//   GEMM1: per wave 32x128 tile of h (16 wmma accum tiles), K = 2048
//   stage h (bf16, fragment-swizzled) in LDS: [mt][ks][plane][lane][8] = 64KB
//   GEMM2: per wave 32x64 tile (8 accum tiles), K = 1024
//   persistent gate-weighted f32 accumulation across the 8 experts
// ---------------------------------------------------------------------------
__global__ void __launch_bounds__(256, 1)
k_moe(const __bf16* __restrict__ Xs,      // fragment-swizzled X
      const __bf16* __restrict__ W1s,     // fragment-swizzled W1
      const __bf16* __restrict__ W2s,     // fragment-swizzled W2
      const float*  __restrict__ gate,    // [B, E]
      const float* __restrict__ b1, const float* __restrict__ g1,
      const float* __restrict__ be1, const float* __restrict__ m1,
      const float* __restrict__ v1,
      const float* __restrict__ b2, const float* __restrict__ g2,
      const float* __restrict__ be2, const float* __restrict__ m2,
      const float* __restrict__ v2,
      float* __restrict__ out) {          // [B, H2]
    extern __shared__ __bf16 hlds[];      // 2 * KS2 * 512 halves = 64 KB

    const int b0   = blockIdx.x * BT;
    const int lane = threadIdx.x & 31;
    const int wave = threadIdx.x >> 5;
    const int lr   = lane & 15;           // column-in-tile (N for B/C/D)
    const int lh   = lane >> 4;           // lane half -> K/M sub-offset

    const v8f vzero = {};
    v8f wsum[8];                          // persistent 32x64 weighted output
#pragma unroll
    for (int i = 0; i < 8; ++i) wsum[i] = vzero;

    for (int e = 0; e < E_; ++e) {
        // ---------------- GEMM1: h = X @ W1[e], wave tile 32 x 128 ----------
        // B blocks for this wave: ntl = wave*8 + nt
        const __bf16* W1e = W1s + (long)(e * (H1_ / 16) + wave * 8) * (KS1 * 512);
        v8f acc1[16];
#pragma unroll
        for (int i = 0; i < 16; ++i) acc1[i] = vzero;

        for (int ks = 0; ks < KS1; ++ks) {
            v16bf afr[2];
#pragma unroll
            for (int mt = 0; mt < 2; ++mt)
                afr[mt] = load_frag_g(
                    Xs + ((long)(blockIdx.x * 2 + mt) * KS1 + ks) * 512, lane);
            v16bf bfr[8];
#pragma unroll
            for (int nt = 0; nt < 8; ++nt)
                bfr[nt] = load_frag_g(W1e + ((long)nt * KS1 + ks) * 512, lane);
#pragma unroll
            for (int mt = 0; mt < 2; ++mt)
#pragma unroll
                for (int nt = 0; nt < 8; ++nt)
                    acc1[mt * 8 + nt] = __builtin_amdgcn_wmma_f32_16x16x32_bf16(
                        false, afr[mt], false, bfr[nt], (short)0,
                        acc1[mt * 8 + nt], false, false);
        }

        // ---------------- BN1 + ReLU -> LDS (fragment-swizzled bf16) --------
        __syncthreads();                  // previous expert's LDS reads done
#pragma unroll
        for (int nt = 0; nt < 8; ++nt) {
            const int n   = wave * 128 + nt * 16 + lr;   // h column = GEMM2 K
            const float s = g1[e * H1_ + n] * rsqrtf(v1[e * H1_ + n] + EPS_);
            const float t = be1[e * H1_ + n] + (b1[e * H1_ + n] - m1[e * H1_ + n]) * s;
            const int ks2 = n >> 5;
            const int cp  = (n >> 4) & 1;                // plane
            const int lhp = (n >> 3) & 1;                // dest lane half
            const int j   = n & 7;                       // element within chunk
#pragma unroll
            for (int mt = 0; mt < 2; ++mt) {
                v8f a = acc1[mt * 8 + nt];
#pragma unroll
                for (int r = 0; r < 8; ++r) {
                    float h = fmaxf(fmaf(a[r], s, t), 0.f);
                    const int lp = lhp * 16 + (r + lh * 8);   // dest lane
                    hlds[(mt * KS2 + ks2) * 512 + cp * 256 + lp * 8 + j] = (__bf16)h;
                }
            }
        }
        __syncthreads();

        // gate values for this lane's output rows
        float gg[16];
#pragma unroll
        for (int mt = 0; mt < 2; ++mt)
#pragma unroll
            for (int r = 0; r < 8; ++r)
                gg[mt * 8 + r] = gate[(long)(b0 + mt * 16 + r + lh * 8) * E_ + e];

        // ---------------- GEMM2: o = h @ W2[e], wave tile 32 x 64 -----------
        const __bf16* W2e = W2s + (long)(e * (H2_ / 16) + wave * 4) * (KS2 * 512);
        v8f acc2[8];
#pragma unroll
        for (int i = 0; i < 8; ++i) acc2[i] = vzero;

        for (int ks = 0; ks < KS2; ++ks) {
            v16bf afr[2];
#pragma unroll
            for (int mt = 0; mt < 2; ++mt)
                afr[mt] = load_frag_l(hlds + (mt * KS2 + ks) * 512, lane);
            v16bf bfr[4];
#pragma unroll
            for (int nt = 0; nt < 4; ++nt)
                bfr[nt] = load_frag_g(W2e + ((long)nt * KS2 + ks) * 512, lane);
#pragma unroll
            for (int mt = 0; mt < 2; ++mt)
#pragma unroll
                for (int nt = 0; nt < 4; ++nt)
                    acc2[mt * 4 + nt] = __builtin_amdgcn_wmma_f32_16x16x32_bf16(
                        false, afr[mt], false, bfr[nt], (short)0,
                        acc2[mt * 4 + nt], false, false);
        }

        // ---------------- BN2 + ReLU + gate-weighted accumulate -------------
#pragma unroll
        for (int nt = 0; nt < 4; ++nt) {
            const int n   = wave * 64 + nt * 16 + lr;
            const float s = g2[e * H2_ + n] * rsqrtf(v2[e * H2_ + n] + EPS_);
            const float t = be2[e * H2_ + n] + (b2[e * H2_ + n] - m2[e * H2_ + n]) * s;
#pragma unroll
            for (int mt = 0; mt < 2; ++mt) {
                v8f a = acc2[mt * 4 + nt];
#pragma unroll
                for (int r = 0; r < 8; ++r) {
                    float o = fmaxf(fmaf(a[r], s, t), 0.f);
                    wsum[mt * 4 + nt][r] = fmaf(gg[mt * 8 + r], o,
                                                wsum[mt * 4 + nt][r]);
                }
            }
        }
    }

    // ---------------- store out tile (each element written once) -----------
#pragma unroll
    for (int mt = 0; mt < 2; ++mt)
#pragma unroll
        for (int nt = 0; nt < 4; ++nt)
#pragma unroll
            for (int r = 0; r < 8; ++r) {
                const int m = mt * 16 + r + lh * 8;
                const int n = wave * 64 + nt * 16 + lr;
                out[(long)(b0 + m) * H2_ + n] = wsum[mt * 4 + nt][r];
            }
}

// ---------------------------------------------------------------------------
// Launch
// ---------------------------------------------------------------------------
extern "C" void kernel_launch(void* const* d_in, const int* in_sizes, int n_in,
                              void* d_out, int out_size, void* d_ws, size_t ws_size,
                              hipStream_t stream) {
    const float* x    = (const float*)d_in[0];
    const float* side = (const float*)d_in[1];
    const float* W1   = (const float*)d_in[2];
    const float* b1   = (const float*)d_in[3];
    const float* g1   = (const float*)d_in[4];
    const float* be1  = (const float*)d_in[5];
    const float* m1   = (const float*)d_in[6];
    const float* v1   = (const float*)d_in[7];
    const float* W2   = (const float*)d_in[8];
    const float* b2   = (const float*)d_in[9];
    const float* g2   = (const float*)d_in[10];
    const float* be2  = (const float*)d_in[11];
    const float* m2   = (const float*)d_in[12];
    const float* v2   = (const float*)d_in[13];
    const float* Wg   = (const float*)d_in[14];
    const float* bg   = (const float*)d_in[15];
    float* out = (float*)d_out;

    // workspace layout (bytes): Xs bf16 | W1s bf16 | W2s bf16 | gate f32
    char* ws = (char*)d_ws;
    __bf16* Xs   = (__bf16*)(ws);                                   // 64 MiB
    __bf16* W1s  = (__bf16*)(ws + (long)B_ * D_ * 2);               // 32 MiB
    __bf16* W2s  = (__bf16*)(ws + (long)B_ * D_ * 2
                                + (long)E_ * D_ * H1_ * 2);         //  8 MiB
    float*  gate = (float*)(ws + (long)B_ * D_ * 2
                               + (long)E_ * D_ * H1_ * 2
                               + (long)E_ * H1_ * H2_ * 2);         // 0.5 MiB

    // 1) pack X: (B/16)*(D/32)*64 chunk-threads
    k_pack_X<<<dim3((B_ / 16) * KS1 * 64 / 256), dim3(256), 0, stream>>>(x, Xs);
    // 2) pack W1: E*(H1/16)*(D/32)*64
    k_pack_W1<<<dim3(E_ * (H1_ / 16) * KS1 * 64 / 256), dim3(256), 0, stream>>>(W1, W1s);
    // 3) pack W2: E*(H2/16)*(H1/32)*64
    k_pack_W2<<<dim3(E_ * (H2_ / 16) * KS2 * 64 / 256), dim3(256), 0, stream>>>(W2, W2s);
    // 4) gate softmax
    k_gate<<<dim3(B_ / 256), dim3(256), 0, stream>>>(side, Wg, bg, gate);
    // 5) fused expert pipeline, dynamic LDS = 64 KiB exact
    k_moe<<<dim3(B_ / BT), dim3(256), 2 * KS2 * 512 * sizeof(__bf16), stream>>>(
        Xs, W1s, W2s, gate,
        b1, g1, be1, m1, v1,
        b2, g2, be2, m2, v2,
        out);
}
